// UpBlock_83640193122774
// MI455X (gfx1250) — matrix operand order repaired and checked by reading, not compiled, not run on hardware
//
#include <hip/hip_runtime.h>
#include <stdint.h>

// ---------------------------------------------------------------------------
// MI455X (gfx1250) implementation.
// All three GEMMs run on v_wmma_f32_16x16x32_bf16 (bf16 in, f32 accumulate).
// Weights are pre-packed into the exact WMMA B-matrix per-lane layout and
// staged in LDS; A tiles stream from global in the striped A layout.
// BN statistics are fused into GEMM epilogues where possible.
// Workspace requirement: ~280 MB.
// ---------------------------------------------------------------------------

typedef __attribute__((ext_vector_type(16))) __bf16 v16bf;
typedef __attribute__((ext_vector_type(8)))  float  v8f;

#define N_IN   40000
#define N_OUT  150000
#define KVOL   8
#define CCH    128
#define EPSBN  1e-5f

union ABits { uint4 q[2]; v16bf v; };

static __device__ __forceinline__ uint16_t f2bf(float f) {
    uint32_t u = __float_as_uint(f);
    u += 0x7FFFu + ((u >> 16) & 1u);          // round-to-nearest-even
    return (uint16_t)(u >> 16);
}
static __device__ __forceinline__ uint32_t pack2bf(float lo, float hi) {
    return (uint32_t)f2bf(lo) | ((uint32_t)f2bf(hi) << 16);
}

// ---------------------------------------------------------------------------
// Utility kernels
// ---------------------------------------------------------------------------
__global__ void zero_f32(float* __restrict__ p, size_t n) {
    size_t t = (size_t)blockIdx.x * blockDim.x + threadIdx.x;
    for (size_t e = t; e < n; e += (size_t)gridDim.x * blockDim.x) p[e] = 0.0f;
}

// f32 -> bf16 pairwise (n2 = #pairs)
__global__ void cvt_bf16(const float* __restrict__ in, uint32_t* __restrict__ out, size_t n2) {
    size_t t = (size_t)blockIdx.x * blockDim.x + threadIdx.x;
    for (size_t e = t; e < n2; e += (size_t)gridDim.x * blockDim.x)
        out[e] = pack2bf(in[2 * e], in[2 * e + 1]);
}

// Pack a row-major [KK,NN] f32 weight into the WMMA bf16 B-matrix per-lane
// layout:  dword index = ((kt*(NN/16)+nt)*32 + lane)*8 + j
//   lane<16 : N = nt*16+lane,      dword j holds K = kt*32 + (j>>2)*16 + 2(j&3), +1
//   lane>=16: N = nt*16+(lane-16), dword j holds K = kt*32 + (j>>2)*16 + 8 + 2(j&3), +1
__global__ void pack_w(const float* __restrict__ W, uint32_t* __restrict__ out,
                       int KK, int NN) {
    int t = blockIdx.x * blockDim.x + threadIdx.x;
    int total = (KK / 32) * (NN / 16) * 32 * 8;
    if (t >= total) return;
    int j    = t & 7;
    int lane = (t >> 3) & 31;
    int blk  = t >> 8;                 // kt*(NN/16) + nt
    int ntiles = NN / 16;
    int nt = blk % ntiles;
    int kt = blk / ntiles;
    int n    = nt * 16 + (lane & 15);
    int half = lane >> 4;
    int k0   = kt * 32 + (j >> 2) * 16 + half * 8 + 2 * (j & 3);
    out[t] = pack2bf(W[(size_t)k0 * NN + n], W[(size_t)(k0 + 1) * NN + n]);
}

// ---------------------------------------------------------------------------
// Scatter GEMM: acc[pair_out[k][p]] += x_bf16[pair_in[k][p]] @ W_up[k]
// grid = (ceil(9375/4), 8); block = 128 (4 waves); one 16-row tile per wave.
// ---------------------------------------------------------------------------
__global__ void __launch_bounds__(128) upconv_wmma(
    const uint16_t* __restrict__ xb,      // [N_IN,128] bf16
    const uint32_t* __restrict__ wup_pk,  // [8][8192] packed dwords
    const int* __restrict__ pin,          // [8][N_OUT]
    const int* __restrict__ pout,         // [8][N_OUT]
    float* __restrict__ acc_out)          // [N_OUT,128] f32
{
    __shared__ uint4 sB[2048];            // 32 KB: packed W_up[k]
    const int k = blockIdx.y;
    const uint4* wsrc = (const uint4*)(wup_pk + (size_t)k * 8192);
    for (int i = threadIdx.x; i < 2048; i += 128) sB[i] = wsrc[i];
    __syncthreads();

    const int wave = threadIdx.x >> 5;
    const int lane = threadIdx.x & 31;
    const int pt   = blockIdx.x * 4 + wave;
    if (pt >= N_OUT / 16) return;         // wave-uniform guard; EXEC stays full

    const int m    = lane & 15;
    const int half = lane >> 4;
    const int* pin_k  = pin  + (size_t)k * N_OUT + pt * 16;
    const int* pout_k = pout + (size_t)k * N_OUT + pt * 16;
    const uint16_t* arow = xb + (size_t)pin_k[m] * CCH;   // gathered row

    v8f acc[8];
    #pragma unroll
    for (int nt = 0; nt < 8; ++nt)
        #pragma unroll
        for (int r = 0; r < 8; ++r) acc[nt][r] = 0.0f;

    #pragma unroll
    for (int kt = 0; kt < 4; ++kt) {
        ABits a;
        a.q[0] = *(const uint4*)(arow + kt * 32 + half * 8);
        a.q[1] = *(const uint4*)(arow + kt * 32 + 16 + half * 8);
        #pragma unroll
        for (int nt = 0; nt < 8; ++nt) {
            ABits b;
            const uint4* bp = &sB[((kt * 8 + nt) * 32 + lane) * 2];
            b.q[0] = bp[0]; b.q[1] = bp[1];
            acc[nt] = __builtin_amdgcn_wmma_f32_16x16x32_bf16(
                false, a.v, false, b.v, (short)0, acc[nt], false, false);
        }
    }

    int orow[8];
    #pragma unroll
    for (int r = 0; r < 8; ++r) orow[r] = pout_k[half * 8 + r];
    const int ncol = lane & 15;
    #pragma unroll
    for (int nt = 0; nt < 8; ++nt)
        #pragma unroll
        for (int r = 0; r < 8; ++r)
            unsafeAtomicAdd(&acc_out[(size_t)orow[r] * CCH + nt * 16 + ncol],
                            acc[nt][r]);
}

// ---------------------------------------------------------------------------
// Dense GEMM + bias, with fused BN column-statistics accumulation.
// Y = A @ Bpk + bias ; bnsum/bnsq += column sums / sums of squares.
// KT = K/32 tiles, LDA = A row stride (bf16 elements).
// ---------------------------------------------------------------------------
template<int KT, int LDA>
__global__ void __launch_bounds__(128) gemm_bn(
    const uint16_t* __restrict__ A,
    const uint32_t* __restrict__ Bpk,
    const float* __restrict__ bias,
    float* __restrict__ Y,
    float* __restrict__ bnsum, float* __restrict__ bnsq)
{
    __shared__ uint4 sB[KT * 8 * 32 * 2];       // KT=8 -> 64 KB, KT=4 -> 32 KB
    constexpr int NB = KT * 8 * 32 * 2;
    for (int i = threadIdx.x; i < NB; i += 128) sB[i] = ((const uint4*)Bpk)[i];
    __syncthreads();

    const int wave = threadIdx.x >> 5;
    const int lane = threadIdx.x & 31;
    const int pt   = blockIdx.x * 4 + wave;
    if (pt >= N_OUT / 16) return;

    const int m    = lane & 15;
    const int half = lane >> 4;
    const uint16_t* arow = A + (size_t)(pt * 16 + m) * LDA;

    v8f acc[8];
    #pragma unroll
    for (int nt = 0; nt < 8; ++nt)
        #pragma unroll
        for (int r = 0; r < 8; ++r) acc[nt][r] = 0.0f;

    #pragma unroll
    for (int kt = 0; kt < KT; ++kt) {
        ABits a;
        a.q[0] = *(const uint4*)(arow + kt * 32 + half * 8);
        a.q[1] = *(const uint4*)(arow + kt * 32 + 16 + half * 8);
        #pragma unroll
        for (int nt = 0; nt < 8; ++nt) {
            ABits b;
            const uint4* bp = &sB[((kt * 8 + nt) * 32 + lane) * 2];
            b.q[0] = bp[0]; b.q[1] = bp[1];
            acc[nt] = __builtin_amdgcn_wmma_f32_16x16x32_bf16(
                false, a.v, false, b.v, (short)0, acc[nt], false, false);
        }
    }

    const int ncol = lane & 15;
    const size_t rowbase = (size_t)(pt * 16 + half * 8) * CCH;
    #pragma unroll
    for (int nt = 0; nt < 8; ++nt) {
        const int col = nt * 16 + ncol;
        const float bv = bias[col];
        float s = 0.0f, s2 = 0.0f;
        #pragma unroll
        for (int r = 0; r < 8; ++r) {
            float y = acc[nt][r] + bv;
            Y[rowbase + (size_t)r * CCH + col] = y;
            s += y; s2 += y * y;
        }
        s  += __shfl_xor(s,  16, 32);   // combine half0/half1 rows, same column
        s2 += __shfl_xor(s2, 16, 32);
        if (lane < 16) {
            unsafeAtomicAdd(&bnsum[col], s);
            unsafeAtomicAdd(&bnsq[col],  s2);
        }
    }
}

// ---------------------------------------------------------------------------
// Column statistics over an [nrows,128] f32 matrix (for BN0: scatter output).
// grid = 256 blocks x 128 threads (thread = column).
// ---------------------------------------------------------------------------
__global__ void colstats(const float* __restrict__ X,
                         float* __restrict__ sum, float* __restrict__ sq, int nrows)
{
    const int col = threadIdx.x;
    int rpb = (nrows + gridDim.x - 1) / gridDim.x;
    int r0 = blockIdx.x * rpb;
    int r1 = r0 + rpb; if (r1 > nrows) r1 = nrows;
    float s = 0.0f, s2 = 0.0f;
    for (int r = r0; r < r1; ++r) {
        float v = X[(size_t)r * CCH + col];
        s += v; s2 += v * v;
    }
    unsafeAtomicAdd(&sum[col], s);
    unsafeAtomicAdd(&sq[col],  s2);
}

__global__ void bn_finalize(const float* __restrict__ sum, const float* __restrict__ sq,
                            const float* __restrict__ g, const float* __restrict__ b,
                            float* __restrict__ scale, float* __restrict__ shift,
                            float inv_n)
{
    int c = threadIdx.x;
    float mn = sum[c] * inv_n;
    float vr = sq[c] * inv_n - mn * mn;
    float sc = g[c] * rsqrtf(vr + EPSBN);
    scale[c] = sc;
    shift[c] = b[c] - mn * sc;
}

// up = relu(bn0(acc)); store f32 up (residual) and pack H = [bf16(up), bf16(skip)]
__global__ void bn0_relu_pack(const float* __restrict__ acc, const float* __restrict__ skip,
                              const float* __restrict__ scale, const float* __restrict__ shift,
                              float* __restrict__ U, uint16_t* __restrict__ H)
{
    size_t t = (size_t)blockIdx.x * blockDim.x + threadIdx.x;
    const size_t total = (size_t)N_OUT * CCH;
    for (size_t e = t; e < total; e += (size_t)gridDim.x * blockDim.x) {
        int col = (int)(e & (CCH - 1));
        size_t row = e >> 7;
        float u = fmaxf(acc[e] * scale[col] + shift[col], 0.0f);
        U[e] = u;
        H[row * 256 + col]       = f2bf(u);
        H[row * 256 + CCH + col] = f2bf(skip[e]);
    }
}

// z = bf16(relu(bn(y)))
__global__ void bn_relu_bf16(const float* __restrict__ Y,
                             const float* __restrict__ scale, const float* __restrict__ shift,
                             uint16_t* __restrict__ Z)
{
    size_t t = (size_t)blockIdx.x * blockDim.x + threadIdx.x;
    const size_t total = (size_t)N_OUT * CCH;
    for (size_t e = t; e < total; e += (size_t)gridDim.x * blockDim.x) {
        int col = (int)(e & (CCH - 1));
        Z[e] = f2bf(fmaxf(Y[e] * scale[col] + shift[col], 0.0f));
    }
}

// out = relu(bn(y2)) + up
__global__ void bn_relu_res(const float* __restrict__ Y2,
                            const float* __restrict__ scale, const float* __restrict__ shift,
                            const float* __restrict__ U, float* __restrict__ out)
{
    size_t t = (size_t)blockIdx.x * blockDim.x + threadIdx.x;
    const size_t total = (size_t)N_OUT * CCH;
    for (size_t e = t; e < total; e += (size_t)gridDim.x * blockDim.x) {
        int col = (int)(e & (CCH - 1));
        out[e] = fmaxf(Y2[e] * scale[col] + shift[col], 0.0f) + U[e];
    }
}

// ---------------------------------------------------------------------------
extern "C" void kernel_launch(void* const* d_in, const int* in_sizes, int n_in,
                              void* d_out, int out_size, void* d_ws, size_t ws_size,
                              hipStream_t stream)
{
    const float* x_feat = (const float*)d_in[0];
    const float* skip   = (const float*)d_in[1];
    const int*   pin    = (const int*)d_in[2];
    const int*   pout   = (const int*)d_in[3];
    const float* W_up   = (const float*)d_in[4];
    const float* g0     = (const float*)d_in[5];
    const float* b0     = (const float*)d_in[6];
    const float* W1     = (const float*)d_in[7];
    const float* b1     = (const float*)d_in[8];
    const float* g1     = (const float*)d_in[9];
    const float* be1    = (const float*)d_in[10];
    const float* W2     = (const float*)d_in[11];
    const float* b2     = (const float*)d_in[12];
    const float* g2     = (const float*)d_in[13];
    const float* be2    = (const float*)d_in[14];
    float* out = (float*)d_out;

    char* ws = (char*)d_ws;
    size_t off = 0;
    auto alloc = [&](size_t bytes) -> void* {
        void* p = ws + off;
        off += (bytes + 255) & ~(size_t)255;
        return p;
    };

    const size_t big = (size_t)N_OUT * CCH * sizeof(float);       // 76.8 MB
    float*    accY1  = (float*)alloc(big);                        // acc, then Y1
    float*    U      = (float*)alloc(big);                        // up (f32)
    void*     HY2    = alloc(big);                                // H bf16 [N,256], then Y2 f32
    uint16_t* Z1     = (uint16_t*)alloc((size_t)N_OUT * CCH * 2);
    uint16_t* Xb     = (uint16_t*)alloc((size_t)N_IN * CCH * 2);
    uint32_t* Wup_pk = (uint32_t*)alloc((size_t)KVOL * 8192 * 4);
    uint32_t* W1_pk  = (uint32_t*)alloc(16384 * 4);
    uint32_t* W2_pk  = (uint32_t*)alloc(8192 * 4);
    float*    st     = (float*)alloc(12 * CCH * sizeof(float));
    float *sum0 = st,        *sq0 = st + 128;
    float *sum1 = st + 256,  *sq1 = st + 384;
    float *sum2 = st + 512,  *sq2 = st + 640;
    float *scale0 = st + 768,  *shift0 = st + 896;
    float *scale1 = st + 1024, *shift1 = st + 1152;
    float *scale2 = st + 1280, *shift2 = st + 1408;

    const float inv_n = 1.0f / (float)N_OUT;

    // 0) zero scatter accumulator + stats accumulators
    zero_f32<<<9375, 256, 0, stream>>>(accY1, (size_t)N_OUT * CCH);
    zero_f32<<<1, 256, 0, stream>>>(st, 6 * 128);

    // 1) precision conversion + weight packing
    cvt_bf16<<<((size_t)N_IN * CCH / 2 + 255) / 256, 256, 0, stream>>>(
        x_feat, (uint32_t*)Xb, (size_t)N_IN * CCH / 2);
    for (int k = 0; k < KVOL; ++k)
        pack_w<<<32, 256, 0, stream>>>(W_up + (size_t)k * CCH * CCH, Wup_pk + (size_t)k * 8192, 128, 128);
    pack_w<<<64, 256, 0, stream>>>(W1, W1_pk, 256, 128);
    pack_w<<<32, 256, 0, stream>>>(W2, W2_pk, 128, 128);

    // 2) sparse inverse conv: gather -> WMMA GEMM -> scatter atomic-add
    dim3 gConv((N_OUT / 16 + 3) / 4, KVOL);
    upconv_wmma<<<gConv, 128, 0, stream>>>(Xb, Wup_pk, pin, pout, accY1);

    // 3) BN0 + ReLU; build H = [up_bf16 | skip_bf16]
    colstats<<<256, 128, 0, stream>>>(accY1, sum0, sq0, N_OUT);
    bn_finalize<<<1, 128, 0, stream>>>(sum0, sq0, g0, b0, scale0, shift0, inv_n);
    bn0_relu_pack<<<18750, 256, 0, stream>>>(accY1, skip, scale0, shift0, U, (uint16_t*)HY2);

    // 4) GEMM1 (Y1 reuses acc slot) with fused BN1 stats
    gemm_bn<8, 256><<<(N_OUT / 16 + 3) / 4, 128, 0, stream>>>(
        (const uint16_t*)HY2, W1_pk, b1, accY1, sum1, sq1);
    bn_finalize<<<1, 128, 0, stream>>>(sum1, sq1, g1, be1, scale1, shift1, inv_n);
    bn_relu_bf16<<<18750, 256, 0, stream>>>(accY1, scale1, shift1, Z1);

    // 5) GEMM2 (Y2 reuses H slot) with fused BN2 stats
    gemm_bn<4, 128><<<(N_OUT / 16 + 3) / 4, 128, 0, stream>>>(
        Z1, W2_pk, b2, (float*)HY2, sum2, sq2);
    bn_finalize<<<1, 128, 0, stream>>>(sum2, sq2, g2, be2, scale2, shift2, inv_n);

    // 6) out = relu(bn2(y2)) + up
    bn_relu_res<<<18750, 256, 0, stream>>>((const float*)HY2, scale2, shift2, U, out);
}
